// ETM_77446850281884
// MI455X (gfx1250) — compile-verified, matrix operand order checked by reading, not run on hardware
//
#include <hip/hip_runtime.h>
#include <math.h>

#define BB 32
#define TT 12
#define KT 50
#define HD 400
#define RH 300
#define V1N 5000
#define V2N 2000
#define VTN 7000
#define GD 1600                       // 4*HD gate width
#define LOG_DELTA (-5.298317367f)     // log(0.005)

typedef __attribute__((ext_vector_type(16))) _Float16 v16h;
typedef __attribute__((ext_vector_type(8)))  float    v8f;

// ---------------------------------------------------------------------------
// Fragment K-index mapping for 16-bit A/B matrices (ISA 7.12.2, 16x32 layout):
// lanes 0-15 hold K = k0+[0..7] and k0+[16..23]; lanes 16-31 the same +8.
// Used only in the K-remainder tail path.
__device__ __forceinline__ int frag_k(int lane, int h) {
  int p = h >> 1;
  int base = (p < 4) ? (p * 2) : (16 + (p - 4) * 2);
  if (lane & 16) base += 8;
  return base + (h & 1);
}

__device__ __forceinline__ v16h cvt_frag(float4 r0, float4 r1, float4 r2, float4 r3) {
  v16h a;
  a[0]  = (_Float16)r0.x; a[1]  = (_Float16)r0.y; a[2]  = (_Float16)r0.z; a[3]  = (_Float16)r0.w;
  a[4]  = (_Float16)r1.x; a[5]  = (_Float16)r1.y; a[6]  = (_Float16)r1.z; a[7]  = (_Float16)r1.w;
  a[8]  = (_Float16)r2.x; a[9]  = (_Float16)r2.y; a[10] = (_Float16)r2.z; a[11] = (_Float16)r2.w;
  a[12] = (_Float16)r3.x; a[13] = (_Float16)r3.y; a[14] = (_Float16)r3.z; a[15] = (_Float16)r3.w;
  return a;
}

// Unguarded fragment load: 4x global_load_b128 (two contiguous 8-float runs).
__device__ __forceinline__ v16h load_frag(const float* __restrict__ row, int k0) {
  const float4* p0 = (const float4*)(row + k0);
  const float4* p1 = (const float4*)(row + k0 + 16);
  return cvt_frag(p0[0], p0[1], p1[0], p1[1]);
}

// C[M,N] = A[M,K] @ Bw[N,K]^T + bias[N]   (PyTorch Linear layout)
// REQUIREMENT: M % 16 == 0 (true for 384 and 32, not for 600 = TT*KT) and all
// operand rows 16B-aligned (lda/ldb multiples of 4 floats, true here).
// One wave per 16x64 output tile; 4 WMMA issues per 32-wide K step. Interior
// blocks (tileN+64 <= N, the common case) run a branchless/maskless K-loop;
// only the single N-edge block column takes the clamped+select path.
__global__ void gemm_wmma_bias(const float* __restrict__ A, int lda,
                               const float* __restrict__ Bw, int ldb,
                               const float* __restrict__ bias,
                               float* __restrict__ C, int ldc,
                               int M, int N, int K)
{
  const int lane  = (int)threadIdx.x & 31;
  const int tileM = blockIdx.y * 16;
  const int tileN = blockIdx.x * 64;
  const int l15   = lane & 15;
  const int koff  = (lane & 16) ? 8 : 0;
  const int mrow  = tileM + l15;

  v8f acc[4] = {v8f{}, v8f{}, v8f{}, v8f{}};

  const int Kmain = K & ~31;
  const float* arow = A + (long)mrow * lda + koff;

  if (tileN + 64 <= N) {
    // ---- interior fast path: no masking at all ----
    const float* brow0 = Bw + (long)(tileN +  0 + l15) * ldb + koff;
    const float* brow1 = Bw + (long)(tileN + 16 + l15) * ldb + koff;
    const float* brow2 = Bw + (long)(tileN + 32 + l15) * ldb + koff;
    const float* brow3 = Bw + (long)(tileN + 48 + l15) * ldb + koff;
    for (int k0 = 0; k0 < Kmain; k0 += 32) {
      if (k0 + 128 < Kmain)
        __builtin_prefetch(arow + k0 + 128, 0, 3);    // global_prefetch_b8
      v16h a  = load_frag(arow,  k0);
      v16h b0 = load_frag(brow0, k0);
      v16h b1 = load_frag(brow1, k0);
      v16h b2 = load_frag(brow2, k0);
      v16h b3 = load_frag(brow3, k0);
      acc[0] = __builtin_amdgcn_wmma_f32_16x16x32_f16(false, a, false, b0, (short)0, acc[0], false, false);
      acc[1] = __builtin_amdgcn_wmma_f32_16x16x32_f16(false, a, false, b1, (short)0, acc[1], false, false);
      acc[2] = __builtin_amdgcn_wmma_f32_16x16x32_f16(false, a, false, b2, (short)0, acc[2], false, false);
      acc[3] = __builtin_amdgcn_wmma_f32_16x16x32_f16(false, a, false, b3, (short)0, acc[3], false, false);
    }
  } else {
    // ---- N-edge path: clamped rows + per-lane select-zero ----
    for (int k0 = 0; k0 < Kmain; k0 += 32) {
      v16h a = load_frag(arow, k0);
#pragma unroll
      for (int j = 0; j < 4; ++j) {
        const int jn = tileN + j * 16;
        if (jn >= N) continue;                        // uniform skip
        int ncol = jn + l15;
        int nc   = ncol < N ? ncol : N - 1;           // clamped, in-bounds
        v16h b = load_frag(Bw + (long)nc * ldb + koff, k0);
        if (ncol >= N) b = v16h{};                    // per-lane zero (VALU)
        acc[j] = __builtin_amdgcn_wmma_f32_16x16x32_f16(
            false, a, false, b, (short)0, acc[j], false, false);
      }
    }
  }

  if (Kmain < K) {                                    // K remainder: once, branchless
    v16h a;
#pragma unroll
    for (int h = 0; h < 16; ++h) {
      int kk = Kmain + frag_k(lane, h);
      int kc = kk < K ? kk : K - 1;
      float av = A[(long)mrow * lda + kc];
      a[h] = (kk < K) ? (_Float16)av : (_Float16)0.0f;
    }
#pragma unroll
    for (int j = 0; j < 4; ++j) {
      const int jn = tileN + j * 16;
      if (jn >= N) continue;
      int ncol = jn + l15;
      int nc   = ncol < N ? ncol : N - 1;
      v16h b;
#pragma unroll
      for (int h = 0; h < 16; ++h) {
        int kk = Kmain + frag_k(lane, h);
        int kc = kk < K ? kk : K - 1;
        float bv = Bw[(long)nc * ldb + kc];
        b[h] = (kk < K) ? (_Float16)bv : (_Float16)0.0f;
      }
      if (ncol >= N) b = v16h{};
      acc[j] = __builtin_amdgcn_wmma_f32_16x16x32_f16(
          false, a, false, b, (short)0, acc[j], false, false);
    }
  }

#pragma unroll
  for (int j = 0; j < 4; ++j) {
    const int jn = tileN + j * 16;
    if (jn >= N) continue;                            // uniform
    int col = jn + l15;
    if (col < N) {
      float bv = bias ? bias[col] : 0.0f;
#pragma unroll
      for (int r = 0; r < 8; ++r) {
        int row = tileM + r + ((lane & 16) ? 8 : 0);  // unguarded: can reach M+7 when M%16 != 0 (M=600)
        C[(long)row * ldc + col] = acc[j][r] + bv;
      }
    }
  }
}

// ---------------------------------------------------------------------------
__global__ void zero_kernel(float* __restrict__ p, long n) {
  long i = (long)blockIdx.x * blockDim.x + threadIdx.x;
  if (i < n) p[i] = 0.0f;
}

__global__ void copy_kernel(float* __restrict__ d, const float* __restrict__ s, long n) {
  long i = (long)blockIdx.x * blockDim.x + threadIdx.x;
  if (i < n) d[i] = s[i];
}

// LSTM cell: gates_x row = b*TT+t (has bih), gates_h row = b (has bhh)
__global__ void lstm_cell(const float* __restrict__ gx, const float* __restrict__ gh,
                          float* __restrict__ h, float* __restrict__ c,
                          float* __restrict__ outx, int t)
{
  int idx = blockIdx.x * blockDim.x + threadIdx.x;
  if (idx >= BB * HD) return;
  int b = idx / HD, j = idx % HD;
  const float* gxr = gx + ((long)b * TT + t) * GD;
  const float* ghr = gh + (long)b * GD;
  float gi = gxr[j]        + ghr[j];
  float gf = gxr[HD + j]   + ghr[HD + j];
  float gg = gxr[2*HD + j] + ghr[2*HD + j];
  float go = gxr[3*HD + j] + ghr[3*HD + j];
  float si = 1.0f / (1.0f + __expf(-gi));
  float sf = 1.0f / (1.0f + __expf(-gf));
  float so = 1.0f / (1.0f + __expf(-go));
  float cn = sf * c[idx] + si * tanhf(gg);
  float hn = so * tanhf(cn);
  c[idx] = cn;
  h[idx] = hn;
  outx[((long)b * TT + t) * HD + j] = hn;
}

// theta heads: mu/ls = [o_t, z_prev] @ W^T + b   (W: [KT, HD+KT])
__global__ void theta_step(const float* __restrict__ out, const float* __restrict__ zprev,
                           const float* __restrict__ muw, const float* __restrict__ mub,
                           const float* __restrict__ lsw, const float* __restrict__ lsb,
                           float* __restrict__ zs_t, float* __restrict__ ls_t, int t)
{
  int idx = blockIdx.x * blockDim.x + threadIdx.x;
  if (idx >= BB * KT) return;
  int b = idx / KT, k = idx % KT;
  const float* o = out + ((long)b * TT + t) * HD;
  float mu = mub[k], ls = lsb[k];
  const int LDW = HD + KT;
  for (int j = 0; j < HD; ++j) {
    float ov = o[j];
    mu += muw[k * LDW + j] * ov;
    ls += lsw[k * LDW + j] * ov;
  }
  if (zprev) {
    for (int j = 0; j < KT; ++j) {
      float zv = zprev[b * KT + j];
      mu += muw[k * LDW + HD + j] * zv;
      ls += lsw[k * LDW + HD + j] * zv;
    }
  }
  zs_t[idx] = mu;
  ls_t[idx] = ls;
}

__global__ void kl_theta_kernel(const float* __restrict__ mu, const float* __restrict__ ls,
                                const float* __restrict__ zprev, float* __restrict__ outp,
                                float pls_const, int use_zprev)
{
  int b = threadIdx.x;
  if (b >= BB) return;
  float s = 0.0f;
  for (int k = 0; k < KT; ++k) {
    float qm = mu[b * KT + k], qls = ls[b * KT + k];
    float pm = use_zprev ? zprev[b * KT + k] : 0.0f;
    float pls = use_zprev ? pls_const : 0.0f;
    float d = qm - pm;
    s += (__expf(qls) + d * d) / (__expf(pls) + 1e-6f) - 1.0f + pls - qls;
  }
  atomicAdd(outp, 0.5f * s / (float)BB);
}

// thread-per-row softmax (short rows)
__global__ void softmax_rows_small(float* __restrict__ x, int rows, int n) {
  int r = blockIdx.x * blockDim.x + threadIdx.x;
  if (r >= rows) return;
  float* p = x + (long)r * n;
  float m = -1e30f;
  for (int i = 0; i < n; ++i) m = fmaxf(m, p[i]);
  float s = 0.0f;
  for (int i = 0; i < n; ++i) { float e = __expf(p[i] - m); p[i] = e; s += e; }
  float inv = 1.0f / s;
  for (int i = 0; i < n; ++i) p[i] *= inv;
}

// block-per-row softmax with LDS reductions (wide rows)
__global__ void softmax_rows_block(float* __restrict__ x, int n) {
  __shared__ float red[256];
  float* p = x + (long)blockIdx.x * n;
  int tid = threadIdx.x;
  float m = -1e30f;
  for (int i = tid; i < n; i += blockDim.x) m = fmaxf(m, p[i]);
  red[tid] = m; __syncthreads();
  for (int s = 128; s > 0; s >>= 1) {
    if (tid < s) red[tid] = fmaxf(red[tid], red[tid + s]);
    __syncthreads();
  }
  m = red[0]; __syncthreads();
  float sum = 0.0f;
  for (int i = tid; i < n; i += blockDim.x) { float e = __expf(p[i] - m); p[i] = e; sum += e; }
  red[tid] = sum; __syncthreads();
  for (int s = 128; s > 0; s >>= 1) {
    if (tid < s) red[tid] += red[tid + s];
    __syncthreads();
  }
  float inv = 1.0f / red[0];
  for (int i = tid; i < n; i += blockDim.x) p[i] *= inv;
}

// alphaT[(t*KT+k)*RH + r] = mu_q_alpha[(k*TT+t)*RH + r]
__global__ void alpha_transpose(const float* __restrict__ mqa, float* __restrict__ alphaT) {
  int idx = blockIdx.x * blockDim.x + threadIdx.x;
  if (idx >= TT * KT * RH) return;
  int r = idx % RH;
  int tk = idx / RH;
  int k = tk % KT, t = tk / KT;
  alphaT[idx] = mqa[((long)k * TT + t) * RH + r];
}

__global__ void kld_alpha_kernel(const float* __restrict__ qmu, const float* __restrict__ qls,
                                 float* __restrict__ outp)
{
  int idx = blockIdx.x * blockDim.x + threadIdx.x;
  if (idx >= TT * KT) return;
  int t = idx / KT, k = idx % KT;
  float s = 0.0f;
  for (int r = 0; r < RH; ++r) {
    float qm = qmu[((long)k * TT + t) * RH + r];
    float ql = qls[((long)k * TT + t) * RH + r];
    float pm = t ? qmu[((long)k * TT + (t - 1)) * RH + r] : 0.0f;
    float pl = t ? LOG_DELTA : 0.0f;
    float d = qm - pm;
    s += (__expf(ql) + d * d) / (__expf(pl) + 1e-6f) - 1.0f + pl - ql;
  }
  atomicAdd(outp, 0.5f * s / (float)KT);
}

__device__ __forceinline__ unsigned hash32(unsigned x) {
  x ^= x >> 16; x *= 0x7feb352dU;
  x ^= x >> 15; x *= 0x846ca68bU;
  x ^= x >> 16; return x;
}

// pass 1: per (b,v): pi=softmax_k(theta*beta), gumbel perturb, argmax -> z stats + kld_z
__global__ void likelihood_pass1(const float* __restrict__ theta_t,
                                 const float* __restrict__ beta_t, int V,
                                 float* __restrict__ zsum0, float* __restrict__ zsum1,
                                 float* __restrict__ kldz_out, unsigned seed)
{
  int idx = blockIdx.x * blockDim.x + threadIdx.x;
  if (idx >= BB * V) return;
  int b = idx / V, v = idx % V;
  const float* th = theta_t + b * KT;

  float m = -1e30f;
  for (int k = 0; k < KT; ++k) m = fmaxf(m, th[k] * beta_t[(long)k * V + v]);
  float den = 0.0f;
  for (int k = 0; k < KT; ++k) den += __expf(th[k] * beta_t[(long)k * V + v] - m);
  float inv = 1.0f / den;

  float best = -1e30f; int kmax = 0; float pimax = 0.0f;
  for (int k = 0; k < KT; ++k) {
    float pi = __expf(th[k] * beta_t[(long)k * V + v] - m) * inv;
    unsigned u = hash32(seed ^ ((unsigned)idx * 53u + (unsigned)k * 0x9e3779b9u));
    float uf = ((float)u + 0.5f) * (1.0f / 4294967296.0f);
    float g = -__logf(-__logf(uf));
    float y = pi + g;
    if (y > best) { best = y; kmax = k; pimax = pi; }
  }
  atomicAdd(&zsum0[(long)v * KT + kmax], 1.0f);
  atomicAdd(&zsum1[(long)b * KT + kmax], 1.0f);
  float contrib = -__logf(fmaxf(th[kmax], 1e-30f)) * (1.0f / ((float)BB * (float)V))
                  - pimax * __logf(fmaxf(pimax, 1e-30f)) * (1.0f / (float)BB);
  atomicAdd(kldz_out, contrib);
}

// pass 2: ll contribution -> d_out[0] accumulates recon = -ll
__global__ void likelihood_pass2(const float* __restrict__ theta_t,
                                 const float* __restrict__ beta_t, int V,
                                 const float* __restrict__ zsum0,
                                 const float* __restrict__ zsum1,
                                 const float* __restrict__ bows, int t, int vofs,
                                 float* __restrict__ recon_out)
{
  int idx = blockIdx.x * blockDim.x + threadIdx.x;
  if (idx >= BB * V) return;
  int b = idx / V, v = idx % V;
  float invV = 1.0f / (float)V, invB = 1.0f / (float)BB;
  float s = 0.0f;
  for (int k = 0; k < KT; ++k) {
    float tz = theta_t[b * KT + k] * (zsum1[b * KT + k] * invV);
    float bz = beta_t[(long)k * V + v] * (zsum0[(long)v * KT + k] * invB);
    s += tz * bz;
  }
  float bw = bows[((long)b * TT + t) * VTN + vofs + v];
  atomicAdd(recon_out, -__logf(fmaxf(s, 1e-30f)) * bw * invB);
}

// ---------------------------------------------------------------------------
static inline void launch_zero(float* p, long n, hipStream_t s) {
  zero_kernel<<<dim3((unsigned)((n + 255) / 256)), dim3(256), 0, s>>>(p, n);
}

static inline void launch_gemm(const float* A, int lda, const float* Bw, int ldb,
                               const float* bias, float* C, int ldc,
                               int M, int N, int K, hipStream_t s) {
  dim3 grid((N + 63) / 64, (M + 15) / 16);
  gemm_wmma_bias<<<grid, dim3(32), 0, s>>>(A, lda, Bw, ldb, bias, C, ldc, M, N, K);
}

extern "C" void kernel_launch(void* const* d_in, const int* in_sizes, int n_in,
                              void* d_out, int out_size, void* d_ws, size_t ws_size,
                              hipStream_t stream)
{
  (void)in_sizes; (void)n_in; (void)out_size; (void)ws_size;
  const float* bows   = (const float*)d_in[0];
  const float* nbows  = (const float*)d_in[1];
  const float* qmap_w = (const float*)d_in[2];
  const float* qmap_b = (const float*)d_in[3];
  const float* wih[2] = {(const float*)d_in[4], (const float*)d_in[8]};
  const float* whh[2] = {(const float*)d_in[5], (const float*)d_in[9]};
  const float* bih[2] = {(const float*)d_in[6], (const float*)d_in[10]};
  const float* bhh[2] = {(const float*)d_in[7], (const float*)d_in[11]};
  const float* muw    = (const float*)d_in[12];
  const float* mub    = (const float*)d_in[13];
  const float* lsw    = (const float*)d_in[14];
  const float* lsb    = (const float*)d_in[15];
  const float* mqa    = (const float*)d_in[16];
  const float* lqa    = (const float*)d_in[17];
  const float* rho1w  = (const float*)d_in[18];
  const float* rho1b  = (const float*)d_in[19];
  const float* rho2w  = (const float*)d_in[20];
  const float* rho2b  = (const float*)d_in[21];
  float* out = (float*)d_out;   // [recon, kld_theta, kld_z1, kld_z2, kld_alpha]

  const int BT = BB * TT;       // 384 rows, row index = b*TT + t

  // workspace carve-up (floats)
  float* W      = (float*)d_ws;
  float* x0     = W;                    // [BT,HD]
  float* gatesx = x0     + (long)BT * HD;       // [BT,GD]
  float* gh     = gatesx + (long)BT * GD;       // [BB,GD]
  float* hbuf   = gh     + (long)BB * GD;       // [BB,HD]
  float* cbuf   = hbuf   + (long)BB * HD;       // [BB,HD]
  float* out0   = cbuf   + (long)BB * HD;       // [BT,HD]
  float* out1   = out0   + (long)BT * HD;       // [BT,HD]
  float* zsb    = out1   + (long)BT * HD;       // [TT,BB,KT]
  float* lsbuf  = zsb    + (long)TT * BB * KT;  // [BB,KT]
  float* theta  = lsbuf  + (long)BB * KT;       // [TT,BB,KT]
  float* alphaT = theta  + (long)TT * BB * KT;  // [TT*KT,RH]
  float* blog1  = alphaT + (long)TT * KT * RH;  // [TT*KT,V1]
  float* blog2  = blog1  + (long)TT * KT * V1N; // [TT*KT,V2]
  float* zsum0  = blog2  + (long)TT * KT * V2N; // [V1,KT] (max)
  float* zsum1  = zsum0  + (long)V1N * KT;      // [BB,KT]

  // outputs start at zero
  launch_zero(out, 5, stream);

  // ---- qmap: inp = nbows @ qmap_w^T + qmap_b  ([384,7000]x[400,7000]) ----
  launch_gemm(nbows, VTN, qmap_w, VTN, qmap_b, x0, HD, BT, HD, VTN, stream);

  // ---- 2-layer LSTM ----
  const float* layer_in = x0;
  float* layer_out = out0;
  for (int L = 0; L < 2; ++L) {
    launch_zero(hbuf, (long)BB * HD * 2, stream);  // h and c (contiguous)
    // batched input gates for all timesteps: [384,400]x[1600,400]
    launch_gemm(layer_in, HD, wih[L], HD, bih[L], gatesx, GD, BT, GD, HD, stream);
    for (int t = 0; t < TT; ++t) {
      launch_gemm(hbuf, HD, whh[L], HD, bhh[L], gh, GD, BB, GD, HD, stream);
      lstm_cell<<<dim3((BB * HD + 255) / 256), dim3(256), 0, stream>>>(
          gatesx, gh, hbuf, cbuf, layer_out, t);
    }
    layer_in = layer_out;
    layer_out = out1;
  }

  // ---- theta recursion + kld_theta ----
  for (int t = 0; t < TT; ++t) {
    const float* zprev = (t == 0) ? nullptr : (zsb + (long)(t - 1) * BB * KT);
    float* zs_t = zsb + (long)t * BB * KT;
    theta_step<<<dim3((BB * KT + 255) / 256), dim3(256), 0, stream>>>(
        out1, zprev, muw, mub, lsw, lsb, zs_t, lsbuf, t);
    kl_theta_kernel<<<dim3(1), dim3(32), 0, stream>>>(
        zs_t, lsbuf, zprev, out + 1, LOG_DELTA, t == 0 ? 0 : 1);
  }
  copy_kernel<<<dim3((TT * BB * KT + 255) / 256), dim3(256), 0, stream>>>(
      theta, zsb, (long)TT * BB * KT);
  softmax_rows_small<<<dim3((TT * BB + 255) / 256), dim3(256), 0, stream>>>(
      theta, TT * BB, KT);

  // ---- alpha -> beta ----
  alpha_transpose<<<dim3((TT * KT * RH + 255) / 256), dim3(256), 0, stream>>>(mqa, alphaT);
  launch_gemm(alphaT, RH, rho1w, RH, rho1b, blog1, V1N, TT * KT, V1N, RH, stream);
  launch_gemm(alphaT, RH, rho2w, RH, rho2b, blog2, V2N, TT * KT, V2N, RH, stream);
  softmax_rows_block<<<dim3(TT * KT), dim3(256), 0, stream>>>(blog1, V1N);
  softmax_rows_block<<<dim3(TT * KT), dim3(256), 0, stream>>>(blog2, V2N);

  // ---- kld_alpha ----
  kld_alpha_kernel<<<dim3((TT * KT + 255) / 256), dim3(256), 0, stream>>>(mqa, lqa, out + 4);

  // ---- decode / likelihood (vocab1 then vocab2) ----
  for (int voc = 0; voc < 2; ++voc) {
    int V = voc ? V2N : V1N;
    float* blog = voc ? blog2 : blog1;
    float* kldz = voc ? (out + 3) : (out + 2);
    int vofs = voc ? (VTN - V2N) : 0;
    for (int t = 0; t < TT; ++t) {
      launch_zero(zsum0, (long)V * KT, stream);
      launch_zero(zsum1, (long)BB * KT, stream);
      const float* th_t = theta + (long)t * BB * KT;
      const float* be_t = blog + (long)t * KT * V;
      unsigned seed = 0x85ebca6bU * (unsigned)(voc + 1) + 0x9e3779b9U * (unsigned)t;
      dim3 grid((BB * V + 255) / 256);
      likelihood_pass1<<<grid, dim3(256), 0, stream>>>(th_t, be_t, V, zsum0, zsum1, kldz, seed);
      likelihood_pass2<<<grid, dim3(256), 0, stream>>>(th_t, be_t, V, zsum0, zsum1,
                                                       bows, t, vofs, out + 0);
    }
  }
}
